// GATLocalizationModel_42193758716069
// MI455X (gfx1250) — matrix operand order repaired and checked by reading, not compiled, not run on hardware
//
#include <hip/hip_runtime.h>
#include <hip/hip_bf16.h>
#include <climits>

typedef __attribute__((ext_vector_type(16))) __bf16 v16bf;
typedef __attribute__((ext_vector_type(8)))  float  v8f;

#define TB 256
#define WAVES_PER_BLOCK 8

// ---------- helpers ----------
__device__ __forceinline__ int f2key(float f) {
  int i = __float_as_int(f);
  return (i >= 0) ? i : (i ^ 0x7fffffff);   // monotone int key for float compare
}
__device__ __forceinline__ float key2f(int i) {
  return __int_as_float((i >= 0) ? i : (i ^ 0x7fffffff));
}

__global__ void fill_f32_kernel(float* __restrict__ p, float v, int n) {
  int t = blockIdx.x * blockDim.x + threadIdx.x;
  if (t < n) p[t] = v;
}
__global__ void fill_i32_kernel(int* __restrict__ p, int v, int n) {
  int t = blockIdx.x * blockDim.x + threadIdx.x;
  if (t < n) p[t] = v;
}

// ---------- dense GEMM: out[M,COUT] = A[M,K] @ W[COUT,K]^T + bias, bf16 WMMA ----------
// One wave per 16x16 output tile. K and COUT are compile-time: all range guards fold to
// constants (or branchless VALU masks), loads are unconditional float2 (global_load_b64),
// stores are 8 straight global_store_b32 with immediate offsets. M must be a multiple of 16.
// Fragment layouts per CDNA5 ISA 7.12.2:
//   A (16x32 bf16): lane half=lane/16, m=lane%16; elem 2v+j -> K = (v<4 ? 8*half+2v : 16+8*half+2(v-4)) + j
//   B (32x16 bf16): lane n=lane%16;              elem 2v+j -> K = (v<4 ? 8*half+2v : 16+8*half+2(v-4)) + j
//   D (16x16 f32):  lane col n=lane%16; vgpr v -> row m = v + 8*half
template<int K, int COUT, int ACT>  // ACT: 0 = none, 1 = relu
__global__ void gemm_bf16_wmma_kernel(const float* __restrict__ A,
                                      const float* __restrict__ W,
                                      const float* __restrict__ bias,
                                      float* __restrict__ out,
                                      int tiles)
{
  constexpr int TILES_N = COUT / 16;
  int wave = (blockIdx.x * blockDim.x + threadIdx.x) >> 5;
  if (wave >= tiles) return;
  int lane = threadIdx.x & 31;
  int tileM = wave / TILES_N;
  int tileN = wave - tileM * TILES_N;
  int half = lane >> 4;          // 0 or 1
  int l15  = lane & 15;
  float hmask = 1.0f - (float)half;   // unused; for K==16 the v>=4 guard zeroes k 16..31

  const float* arow = A + (size_t)(tileM * 16 + l15) * K;
  const float* wrow = W + (size_t)(tileN * 16 + l15) * K;

  v8f acc = {};
#pragma unroll
  for (int k0 = 0; k0 < K; k0 += 32) {
    v16bf af, bf;
#pragma unroll
    for (int v = 0; v < 8; ++v) {
      // ---- A fragment: K index = k0 + half*8 + kao (+j) ----
      const int kao = (v < 4) ? (2 * v) : (16 + 2 * (v - 4));
      if (k0 + kao + 8 + 1 < K + 1) {            // worst-case half=1; compile-time exact here
        float2 a2 = *(const float2*)(arow + k0 + half * 8 + kao);
        af[2 * v]     = (__bf16)a2.x;
        af[2 * v + 1] = (__bf16)a2.y;
      } else {
        af[2 * v]     = (__bf16)0.0f;
        af[2 * v + 1] = (__bf16)0.0f;
      }

      // ---- B fragment: K index = k0 + half*8 + kao (+j) ----
      if (k0 + kao + 8 + 1 < K + 1) {
        float2 b2 = *(const float2*)(wrow + k0 + half * 8 + kao);
        bf[2 * v]     = (__bf16)b2.x;
        bf[2 * v + 1] = (__bf16)b2.y;
      } else {
        bf[2 * v]     = (__bf16)0.0f;
        bf[2 * v + 1] = (__bf16)0.0f;
      }
    }
    acc = __builtin_amdgcn_wmma_f32_16x16x32_bf16(false, af, false, bf,
                                                  (short)0, acc, false, false);
  }

  int col = tileN * 16 + l15;
  float bs = bias[col];
  // row for vgpr v is tileM*16 + 8*half + v ; COUT constexpr -> immediate store offsets
  float* orow = out + (size_t)(tileM * 16 + 8 * half) * COUT + col;
#pragma unroll
  for (int v = 0; v < 8; ++v) {
    float r = acc[v] + bs;
    if (ACT == 1) r = fmaxf(r, 0.0f);
    orow[v * COUT] = r;
  }
}

// ---------- edge phase (wave per edge, 32 lanes x 4 channels = 128 wide) ----------
template<int HEADS, int CH>
__global__ void edge_logits_max_kernel(const int* __restrict__ srcI, const int* __restrict__ dstI,
                                       const float* __restrict__ eattr,
                                       const float* __restrict__ xl, const float* __restrict__ xr,
                                       const float* __restrict__ We, const float* __restrict__ att,
                                       float* __restrict__ logits, int* __restrict__ lmaxkey, int E)
{
  const int F = HEADS * CH;   // 128
  int wid = (blockIdx.x * blockDim.x + threadIdx.x) >> 5;
  if (wid >= E) return;
  int lane = threadIdx.x & 31;
  int s = srcI[wid], d = dstI[wid];
  float ea = eattr[wid];
  int cg = lane * 4;

  float4 xv = *(const float4*)(xl  + (size_t)s * F + cg);
  float4 rv = *(const float4*)(xr  + (size_t)d * F + cg);
  float4 wv = *(const float4*)(We  + cg);
  float4 av = *(const float4*)(att + cg);

  float m0 = xv.x + rv.x + ea * wv.x; m0 = (m0 > 0.f) ? m0 : 0.2f * m0;
  float m1 = xv.y + rv.y + ea * wv.y; m1 = (m1 > 0.f) ? m1 : 0.2f * m1;
  float m2 = xv.z + rv.z + ea * wv.z; m2 = (m2 > 0.f) ? m2 : 0.2f * m2;
  float m3 = xv.w + rv.w + ea * wv.w; m3 = (m3 > 0.f) ? m3 : 0.2f * m3;
  float partial = m0 * av.x + m1 * av.y + m2 * av.z + m3 * av.w;

  const int GRP = CH / 4;     // lanes per head (8 or 32)
#pragma unroll
  for (int off = GRP >> 1; off > 0; off >>= 1)
    partial += __shfl_xor(partial, off, 32);

  int h = cg / CH;
  if ((lane & (GRP - 1)) == 0) {
    logits[(size_t)wid * HEADS + h] = partial;
    atomicMax(lmaxkey + d * HEADS + h, f2key(partial));
  }
}

template<int HEADS>
__global__ void edge_exp_sum_kernel(const int* __restrict__ dstI, const float* __restrict__ logits,
                                    const int* __restrict__ lmaxkey, float* __restrict__ ex,
                                    float* __restrict__ denom, int E)
{
  int t = blockIdx.x * blockDim.x + threadIdx.x;
  if (t >= E * HEADS) return;
  int e = t / HEADS, h = t - e * HEADS;
  int d = dstI[e];
  float lm = key2f(lmaxkey[d * HEADS + h]);
  float v = __expf(logits[t] - lm);
  ex[t] = v;
  atomicAdd(denom + d * HEADS + h, v);
}

template<int HEADS, int CH>
__global__ void edge_scatter_kernel(const int* __restrict__ srcI, const int* __restrict__ dstI,
                                    const float* __restrict__ ex, const float* __restrict__ denom,
                                    const float* __restrict__ xl, float* __restrict__ acc, int E)
{
  const int F = HEADS * CH;
  int wid = (blockIdx.x * blockDim.x + threadIdx.x) >> 5;
  if (wid >= E) return;
  int lane = threadIdx.x & 31;
  int s = srcI[wid], d = dstI[wid];
  int cg = lane * 4;
  int h = cg / CH;
  float alpha = ex[(size_t)wid * HEADS + h] / denom[d * HEADS + h];
  float4 xv = *(const float4*)(xl + (size_t)s * F + cg);
  float* o = acc + (size_t)d * F + cg;
  atomicAdd(o + 0, alpha * xv.x);
  atomicAdd(o + 1, alpha * xv.y);
  atomicAdd(o + 2, alpha * xv.z);
  atomicAdd(o + 3, alpha * xv.w);
}

__global__ void bias_elu_kernel(float* __restrict__ a, const float* __restrict__ bias, int n, int F) {
  int t = blockIdx.x * blockDim.x + threadIdx.x;
  if (t >= n) return;
  int f = t % F;
  float v = a[t] + bias[f];
  a[t] = (v > 0.f) ? v : (__expf(v) - 1.0f);
}

// final 32 -> 2 head, one thread per node
__global__ void head_out_kernel(const float* __restrict__ h, const float* __restrict__ W3,
                                const float* __restrict__ c3, float* __restrict__ out, int N) {
  int n = blockIdx.x * blockDim.x + threadIdx.x;
  if (n >= N) return;
  const float* hr = h + (size_t)n * 32;
  float s0 = c3[0], s1 = c3[1];
#pragma unroll
  for (int k = 0; k < 32; ++k) {
    float v = hr[k];
    s0 += v * W3[k];
    s1 += v * W3[32 + k];
  }
  out[(size_t)n * 2 + 0] = s0;
  out[(size_t)n * 2 + 1] = s1;
}

// ---------- host side ----------
template<int K, int COUT, int ACT>
static void run_gemm_t(const float* A, const float* W, const float* b, float* out,
                       int M, hipStream_t s) {
  int tiles  = (M / 16) * (COUT / 16);
  int blocks = (tiles + WAVES_PER_BLOCK - 1) / WAVES_PER_BLOCK;
  gemm_bf16_wmma_kernel<K, COUT, ACT><<<blocks, TB, 0, s>>>(A, W, b, out, tiles);
}

extern "C" void kernel_launch(void* const* d_in, const int* in_sizes, int n_in,
                              void* d_out, int out_size, void* d_ws, size_t ws_size,
                              hipStream_t stream) {
  const float* x    = (const float*)d_in[0];
  const int*   eidx = (const int*)  d_in[1];
  const float* eatt = (const float*)d_in[2];
  const float* Wl1  = (const float*)d_in[3];
  const float* bl1  = (const float*)d_in[4];
  const float* Wr1  = (const float*)d_in[5];
  const float* br1  = (const float*)d_in[6];
  const float* We1  = (const float*)d_in[7];
  const float* att1 = (const float*)d_in[8];
  const float* b1   = (const float*)d_in[9];
  const float* Wl2  = (const float*)d_in[10];
  const float* bl2  = (const float*)d_in[11];
  const float* Wr2  = (const float*)d_in[12];
  const float* br2  = (const float*)d_in[13];
  const float* We2  = (const float*)d_in[14];
  const float* att2 = (const float*)d_in[15];
  const float* b2   = (const float*)d_in[16];
  const float* W1   = (const float*)d_in[17];
  const float* c1   = (const float*)d_in[18];
  const float* W2   = (const float*)d_in[19];
  const float* c2   = (const float*)d_in[20];
  const float* W3   = (const float*)d_in[21];
  const float* c3   = (const float*)d_in[22];

  const int N = in_sizes[0] / 16;   // 50000 (divisible by 16)
  const int E = in_sizes[1] / 2;    // 800000
  const int F = 128;
  const int* srcI = eidx;
  const int* dstI = eidx + E;

  // workspace layout
  char* w = (char*)d_ws;
  size_t off = 0;
  auto carve = [&](size_t bytes) -> char* {
    char* p = w + off;
    off = (off + bytes + 255) & ~(size_t)255;
    return p;
  };
  float* xl      = (float*)carve((size_t)N * F * 4);  // reused as xl1/xl2/mlp-h1
  float* xr      = (float*)carve((size_t)N * F * 4);  // reused as xr1/xr2/mlp-h2
  float* h1      = (float*)carve((size_t)N * F * 4);  // layer-1 output
  float* h2      = (float*)carve((size_t)N * F * 4);  // layer-2 output
  float* logits  = (float*)carve((size_t)E * 4 * 4);
  float* ex      = (float*)carve((size_t)E * 4 * 4);
  int*   lmaxkey = (int*)  carve((size_t)N * 4 * 4);
  float* denom   = (float*)carve((size_t)N * 4 * 4);

  const int ewaves_blocks = (E * 32 + TB - 1) / TB;   // wave-per-edge kernels
  auto gb = [](int n) { return (n + TB - 1) / TB; };

  // ===== GATv2 layer 1: in=16 -> 4 heads x 32 =====
  run_gemm_t<16, 128, 0>(x, Wl1, bl1, xl, N, stream);
  run_gemm_t<16, 128, 0>(x, Wr1, br1, xr, N, stream);
  fill_i32_kernel<<<gb(N * 4), TB, 0, stream>>>(lmaxkey, INT_MIN, N * 4);
  fill_f32_kernel<<<gb(N * 4), TB, 0, stream>>>(denom, 0.0f, N * 4);
  fill_f32_kernel<<<gb(N * F), TB, 0, stream>>>(h1, 0.0f, N * F);
  edge_logits_max_kernel<4, 32><<<ewaves_blocks, TB, 0, stream>>>(
      srcI, dstI, eatt, xl, xr, We1, att1, logits, lmaxkey, E);
  edge_exp_sum_kernel<4><<<gb(E * 4), TB, 0, stream>>>(dstI, logits, lmaxkey, ex, denom, E);
  edge_scatter_kernel<4, 32><<<ewaves_blocks, TB, 0, stream>>>(srcI, dstI, ex, denom, xl, h1, E);
  bias_elu_kernel<<<gb(N * F), TB, 0, stream>>>(h1, b1, N * F, F);

  // ===== GATv2 layer 2: in=128 -> 1 head x 128 =====
  run_gemm_t<128, 128, 0>(h1, Wl2, bl2, xl, N, stream);
  run_gemm_t<128, 128, 0>(h1, Wr2, br2, xr, N, stream);
  fill_i32_kernel<<<gb(N), TB, 0, stream>>>(lmaxkey, INT_MIN, N);
  fill_f32_kernel<<<gb(N), TB, 0, stream>>>(denom, 0.0f, N);
  fill_f32_kernel<<<gb(N * F), TB, 0, stream>>>(h2, 0.0f, N * F);
  edge_logits_max_kernel<1, 128><<<ewaves_blocks, TB, 0, stream>>>(
      srcI, dstI, eatt, xl, xr, We2, att2, logits, lmaxkey, E);
  edge_exp_sum_kernel<1><<<gb(E), TB, 0, stream>>>(dstI, logits, lmaxkey, ex, denom, E);
  edge_scatter_kernel<1, 128><<<ewaves_blocks, TB, 0, stream>>>(srcI, dstI, ex, denom, xl, h2, E);
  bias_elu_kernel<<<gb(N * F), TB, 0, stream>>>(h2, b2, N * F, F);

  // ===== MLP head: 128 -> 32 -> 32 -> 2 =====
  run_gemm_t<128, 32, 1>(h2, W1, c1, xl, N, stream);   // relu
  run_gemm_t<32, 32, 1>(xl, W2, c2, xr, N, stream);    // relu
  head_out_kernel<<<gb(N), TB, 0, stream>>>(xr, W3, c3, (float*)d_out, N);
}
